// AttentiveFPReadout_12429635354788
// MI455X (gfx1250) — compile-verified, matrix operand order checked by reading, not run on hardware
//
#include <hip/hip_runtime.h>
#include <stdint.h>

// Problem constants (match reference)
#define GB   2048
#define GS   48
#define GD   256
#define GH   8
#define GDH  32
#define MPAD 64            // x tile rows padded 49 -> 64 for 16x16 WMMA tiling
#define NEG_SLOPE 0.2f

typedef __attribute__((ext_vector_type(16))) __bf16 v16bf;
typedef __attribute__((ext_vector_type(8)))  float  v8f;
typedef unsigned int u32x4 __attribute__((ext_vector_type(4)));
typedef int          i32x8 __attribute__((ext_vector_type(8)));
typedef int          i32x4 __attribute__((ext_vector_type(4)));

__device__ __forceinline__ unsigned short f2bf(float f) {
  unsigned u = __float_as_uint(f);
  u += 0x7FFFu + ((u >> 16) & 1u);          // round-to-nearest-even
  return (unsigned short)(u >> 16);
}
__device__ __forceinline__ float bf2f(unsigned short h) {
  return __uint_as_float(((unsigned)h) << 16);
}

union FragU { v16bf v; uint4 u[2]; };

// Load a 16x32 bf16 WMMA fragment (A-layout; weights are stored N-major so the
// same routine serves B fragments). row-major source, ld elements per row.
// Lane: row = lane&15 (+tile), half = lane>>4 selects K groups {0..7,16..23}+half*8.
__device__ __forceinline__ v16bf load_frag(const unsigned short* base, int row, int k0, int ld) {
  FragU f;
  const unsigned short* p = base + row * ld + k0;
  f.u[0] = *(const uint4*)(p);        // k0 .. k0+7
  f.u[1] = *(const uint4*)(p + 16);   // k0+16 .. k0+23
  return f.v;
}

// ---------------------------------------------------------------------------
// Prep: convert+transpose a KxN f32 weight matrix into NxK bf16 (N-major).
// ---------------------------------------------------------------------------
__global__ void k_transpose_bf16(const float* __restrict__ src,
                                 unsigned short* __restrict__ dst, int K, int N) {
  int idx = blockIdx.x * blockDim.x + threadIdx.x;
  if (idx >= K * N) return;
  int k = idx / N, n = idx - k * N;
  dst[n * K + k] = f2bf(src[idx]);
}

// ---------------------------------------------------------------------------
// Prep: build x = [virtual; nodes] as bf16, and f32 initial state = virtual.
// One block per graph, thread d owns feature column d (coalesced).
// ---------------------------------------------------------------------------
__global__ __launch_bounds__(256) void k_prep_x(const float* __restrict__ nf,
                                                unsigned short* __restrict__ xg,
                                                float* __restrict__ state) {
  const int b = blockIdx.x, d = threadIdx.x;
  const float* p = nf + (size_t)b * GS * GD + d;
  unsigned short* xb = xg + (size_t)b * 49 * GD;
  float v = 0.f;
  for (int s = 0; s < GS; ++s) {
    float f = p[(size_t)s * GD];
    v += f;
    xb[(1 + s) * GD + d] = f2bf(f);
  }
  xb[d] = f2bf(v);
  state[(size_t)b * GD + d] = v;
}

// ---------------------------------------------------------------------------
// GAT step: one block per graph (8 waves = 8 heads).
//   h = x @ Wg            (bf16 WMMA, f32 acc -> bf16 in LDS)
//   e/softmax/msg         (wave-private per head, shuffle reductions)
//   out = relu(x @ Wself + [msg;0])   rows 1..48 -> x (bf16), row 0 -> gruin
// ---------------------------------------------------------------------------
__global__ __launch_bounds__(256) void k_gat_step(
    unsigned short* __restrict__ xg,         // [B][49][256] bf16 (in/out)
    const unsigned short* __restrict__ wgT,  // [256][256] bf16, N-major
    const unsigned short* __restrict__ wsT,  // [256][256] bf16, N-major
    const float* __restrict__ att_src,       // [8][32]
    const float* __restrict__ att_dst,       // [8][32]
    float* __restrict__ gruin)               // [B][256] f32
{
  __shared__ unsigned short xA[MPAD * GD];   // 32 KB  x tile (rows 49..63 zero)
  __shared__ unsigned short hS[49 * GD];     // 24.5 KB h tile (bf16)
  __shared__ float eS[GH * GS];
  __shared__ float alphaS[GH * GS];

  const int b = blockIdx.x;
  const int tid = threadIdx.x;
  const int wv = tid >> 5;        // wave == head
  const int lane = tid & 31;
  const int rA = lane & 15;
  const int half = lane >> 4;

  unsigned short* xgb = xg + (size_t)b * 49 * GD;

  // ---- bring 49x256 bf16 x tile into LDS via the Tensor Data Mover ----
#if __has_builtin(__builtin_amdgcn_tensor_load_to_lds)
  if (tid == 0) {
    unsigned lds_off = (unsigned)(uintptr_t)(&xA[0]);    // LDS aperture: low 32 bits = offset
    unsigned long long ga = (unsigned long long)(uintptr_t)xgb;
    u32x4 g0;
    g0.x = 1u;                                           // count = 1 (valid D#)
    g0.y = lds_off;                                      // lds_addr (bytes)
    g0.z = (unsigned)(ga & 0xFFFFFFFFull);               // global_addr[31:0]
    g0.w = (unsigned)((ga >> 32) & 0x1FFFFFFull) | (2u << 30);  // addr[56:32] | type=2
    i32x8 g1;
    g1[0] = (int)(1u << 16);          // data_size = 1 (2 bytes/elem)
    g1[1] = (int)(256u << 16);        // tensor_dim0 = 256 (bits 79:48, low half)
    g1[2] = (int)(49u << 16);         // tensor_dim1 = 49  (bits 111:80, low half)
    g1[3] = (int)(256u << 16);        // tile_dim0 = 256   (bits 127:112)
    g1[4] = 49;                       // tile_dim1 = 49, tile_dim2 = 0
    g1[5] = 256;                      // tensor_dim0_stride = 256 (elements)
    g1[6] = 0;
    g1[7] = 0;
    i32x4 gz4 = {0, 0, 0, 0};         // <=2D tensor: groups 2/3 unused
    i32x8 gz8 = {0, 0, 0, 0, 0, 0, 0, 0};
    // amdgpu-toolchain (clang-23) 6-arg form: (g0, g1, g2, g3, g_extra, cpol)
    __builtin_amdgcn_tensor_load_to_lds(g0, g1, gz4, gz4, gz8, 0);
  }
  __builtin_amdgcn_s_wait_tensorcnt(0);
#else
  for (int i = tid; i < 49 * GD / 8; i += 256)
    ((uint4*)xA)[i] = ((const uint4*)xgb)[i];
#endif
  for (int i = tid; i < (MPAD - 49) * GD / 2; i += 256)
    ((unsigned*)xA)[49 * GD / 2 + i] = 0u;
  __syncthreads();

  const v8f zf = {0.f, 0.f, 0.f, 0.f, 0.f, 0.f, 0.f, 0.f};

  // ---- matmul 1: h = x @ Wg (wave w produces columns [32w, 32w+32)) ----
  for (int nn = 0; nn < 2; ++nn) {
    const int nt = wv * 2 + nn;
    v8f acc[4];
#pragma unroll
    for (int mt = 0; mt < 4; ++mt) acc[mt] = zf;
    for (int kt = 0; kt < 8; ++kt) {
      const int k0 = kt * 32 + half * 8;
      v16bf bf = load_frag(wgT, nt * 16 + rA, k0, GD);         // global (L2)
#pragma unroll
      for (int mt = 0; mt < 4; ++mt) {
        v16bf af = load_frag(xA, mt * 16 + rA, k0, GD);        // LDS
        acc[mt] = __builtin_amdgcn_wmma_f32_16x16x32_bf16(
            false, af, false, bf, (short)0, acc[mt], false, false);
      }
    }
    const int n = nt * 16 + rA;
#pragma unroll
    for (int mt = 0; mt < 4; ++mt) {
      const int mbase = mt * 16 + half * 8;
#pragma unroll
      for (int i = 0; i < 8; ++i) {
        const int m = mbase + i;
        if (m < 49) hS[m * GD + n] = f2bf(acc[mt][i]);
      }
    }
  }

  // ---- attention for head wv (all deps wave-private; no block barrier) ----
  const float as_l = att_src[wv * GDH + lane];
  const float ad_l = att_dst[wv * GDH + lane];
  float pd = bf2f(hS[wv * GDH + lane]) * ad_l;                 // h row 0 (virtual)
#pragma unroll
  for (int off = 16; off; off >>= 1) pd += __shfl_xor(pd, off, 32);
  const float ed = pd;

  for (int s = 0; s < GS; ++s) {
    float v = bf2f(hS[(1 + s) * GD + wv * GDH + lane]) * as_l;
#pragma unroll
    for (int off = 16; off; off >>= 1) v += __shfl_xor(v, off, 32);
    if (lane == 0) {
      float e = v + ed;
      eS[wv * GS + s] = (e > 0.f) ? e : NEG_SLOPE * e;
    }
  }
  {  // softmax over 48 within the wave
    float v1 = eS[wv * GS + lane];
    float v2 = (lane < 16) ? eS[wv * GS + 32 + lane] : -3.0e38f;
    float mx = fmaxf(v1, v2);
#pragma unroll
    for (int off = 16; off; off >>= 1) mx = fmaxf(mx, __shfl_xor(mx, off, 32));
    float ex1 = __expf(v1 - mx);
    float ex2 = (lane < 16) ? __expf(v2 - mx) : 0.f;
    float sm = ex1 + ex2;
#pragma unroll
    for (int off = 16; off; off >>= 1) sm += __shfl_xor(sm, off, 32);
    float inv = 1.0f / sm;
    alphaS[wv * GS + lane] = ex1 * inv;
    if (lane < 16) alphaS[wv * GS + 32 + lane] = ex2 * inv;
  }
  // msg for head wv: lane owns dim d = wv*32 + lane
  float msgv = 0.f;
  for (int s = 0; s < GS; ++s)
    msgv += alphaS[wv * GS + s] * bf2f(hS[(1 + s) * GD + wv * GDH + lane]);

  // ---- matmul 2: out = relu(x @ Wself (+msg at row 0)) -> global ----
  for (int nn = 0; nn < 2; ++nn) {
    const int nt = wv * 2 + nn;
    v8f acc[4];
#pragma unroll
    for (int mt = 0; mt < 4; ++mt) acc[mt] = zf;
    for (int kt = 0; kt < 8; ++kt) {
      const int k0 = kt * 32 + half * 8;
      v16bf bf = load_frag(wsT, nt * 16 + rA, k0, GD);
#pragma unroll
      for (int mt = 0; mt < 4; ++mt) {
        v16bf af = load_frag(xA, mt * 16 + rA, k0, GD);
        acc[mt] = __builtin_amdgcn_wmma_f32_16x16x32_bf16(
            false, af, false, bf, (short)0, acc[mt], false, false);
      }
    }
    const int n = nt * 16 + rA;
    const float msh = __shfl(msgv, nn ? (rA + 16) : rA, 32);
#pragma unroll
    for (int mt = 0; mt < 4; ++mt) {
      const int mbase = mt * 16 + half * 8;
#pragma unroll
      for (int i = 0; i < 8; ++i) {
        const int m = mbase + i;
        if (m >= 49) continue;
        float v = acc[mt][i];
        if (m == 0) v += msh;
        v = fmaxf(v, 0.f);
        if (m == 0) gruin[(size_t)b * GD + n] = v;            // GRU input (f32)
        else        xgb[m * GD + n] = f2bf(v);                // next-step x
      }
    }
  }
}

// ---------------------------------------------------------------------------
// Batched GRU: 16 batch rows per block; 6 WMMA gemms (z/r/h x wx/wh) per tile.
// ---------------------------------------------------------------------------
__global__ __launch_bounds__(256) void k_gru(
    const float* __restrict__ gruin, float* __restrict__ state,
    const unsigned short* __restrict__ wxT,  // [768][256] bf16, N-major
    const unsigned short* __restrict__ whT,
    const float* __restrict__ bx, const float* __restrict__ bh,
    unsigned short* __restrict__ xg)
{
  __shared__ unsigned short xa[16 * GD];
  __shared__ unsigned short ha[16 * GD];
  __shared__ float st[16 * GD];
  const int tid = threadIdx.x, wv = tid >> 5, lane = tid & 31;
  const int rA = lane & 15, half = lane >> 4;
  const int b0 = blockIdx.x * 16;

  for (int i = tid; i < 16 * GD; i += 256) {
    int r = i >> 8, d = i & 255;
    float xv = gruin[(size_t)(b0 + r) * GD + d];
    float hv = state[(size_t)(b0 + r) * GD + d];
    xa[i] = f2bf(xv); ha[i] = f2bf(hv); st[i] = hv;
  }
  __syncthreads();

  const v8f zf = {0.f, 0.f, 0.f, 0.f, 0.f, 0.f, 0.f, 0.f};
  for (int jj = 0; jj < 2; ++jj) {
    const int j = wv * 2 + jj;           // output 16-col tile within [0,256)
    v8f ax[3], ah[3];
#pragma unroll
    for (int g = 0; g < 3; ++g) { ax[g] = zf; ah[g] = zf; }
    for (int kt = 0; kt < 8; ++kt) {
      const int k0 = kt * 32 + half * 8;
      v16bf afx = load_frag(xa, rA, k0, GD);
      v16bf afh = load_frag(ha, rA, k0, GD);
#pragma unroll
      for (int g = 0; g < 3; ++g) {
        v16bf bfx = load_frag(wxT, g * GD + j * 16 + rA, k0, GD);
        ax[g] = __builtin_amdgcn_wmma_f32_16x16x32_bf16(
            false, afx, false, bfx, (short)0, ax[g], false, false);
        v16bf bfh = load_frag(whT, g * GD + j * 16 + rA, k0, GD);
        ah[g] = __builtin_amdgcn_wmma_f32_16x16x32_bf16(
            false, afh, false, bfh, (short)0, ah[g], false, false);
      }
    }
    const int d = j * 16 + rA;
    const float bxz = bx[d],          bhz = bh[d];
    const float bxr = bx[GD + d],     bhr = bh[GD + d];
    const float bxh = bx[2 * GD + d], bhh = bh[2 * GD + d];
#pragma unroll
    for (int i = 0; i < 8; ++i) {
      const int m = half * 8 + i;
      float z = 1.f / (1.f + __expf(-(ax[0][i] + ah[0][i] + bxz + bhz)));
      float r = 1.f / (1.f + __expf(-(ax[1][i] + ah[1][i] + bxr + bhr)));
      float n = tanhf(ax[2][i] + bxh + r * (ah[2][i] + bhh));
      float hp = st[m * GD + d];
      float ns = z * hp + (1.f - z) * n;
      state[(size_t)(b0 + m) * GD + d] = ns;
      xg[(size_t)(b0 + m) * 49 * GD + d] = f2bf(ns);   // x row 0 = new state
    }
  }
}

// ---------------------------------------------------------------------------
// Final projection: out = state @ proj_w + proj_b
// ---------------------------------------------------------------------------
__global__ __launch_bounds__(256) void k_proj(
    const float* __restrict__ state,
    const unsigned short* __restrict__ pT,   // [256][256] bf16, N-major
    const float* __restrict__ pb, float* __restrict__ out)
{
  __shared__ unsigned short sa[16 * GD];
  const int tid = threadIdx.x, wv = tid >> 5, lane = tid & 31;
  const int rA = lane & 15, half = lane >> 4;
  const int b0 = blockIdx.x * 16;
  for (int i = tid; i < 16 * GD; i += 256)
    sa[i] = f2bf(state[(size_t)b0 * GD + i]);
  __syncthreads();

  const v8f zf = {0.f, 0.f, 0.f, 0.f, 0.f, 0.f, 0.f, 0.f};
  for (int nn = 0; nn < 2; ++nn) {
    const int nt = wv * 2 + nn;
    v8f acc = zf;
    for (int kt = 0; kt < 8; ++kt) {
      const int k0 = kt * 32 + half * 8;
      v16bf af = load_frag(sa, rA, k0, GD);
      v16bf bf = load_frag(pT, nt * 16 + rA, k0, GD);
      acc = __builtin_amdgcn_wmma_f32_16x16x32_bf16(
          false, af, false, bf, (short)0, acc, false, false);
    }
    const int d = nt * 16 + rA;
    const float bias = pb[d];
#pragma unroll
    for (int i = 0; i < 8; ++i) {
      const int m = half * 8 + i;
      out[(size_t)(b0 + m) * GD + d] = acc[i] + bias;
    }
  }
}

// ---------------------------------------------------------------------------
extern "C" void kernel_launch(void* const* d_in, const int* in_sizes, int n_in,
                              void* d_out, int out_size, void* d_ws, size_t ws_size,
                              hipStream_t stream)
{
  const float* node_feature = (const float*)d_in[0];
  // d_in[1] = graph_indicator (unused; graphs are contiguous blocks of 48)
  const float* gat_kernel = (const float*)d_in[2];
  const float* gat_self   = (const float*)d_in[3];
  const float* att_src    = (const float*)d_in[4];
  const float* att_dst    = (const float*)d_in[5];
  const float* gru_wx     = (const float*)d_in[6];
  const float* gru_wh     = (const float*)d_in[7];
  const float* gru_bx     = (const float*)d_in[8];
  const float* gru_bh     = (const float*)d_in[9];
  const float* proj_w     = (const float*)d_in[10];
  const float* proj_b     = (const float*)d_in[11];

  // Workspace layout (~56.8 MB total)
  char* w = (char*)d_ws;
  unsigned short* wgT = (unsigned short*)(w + 0);          // 131072 B
  unsigned short* wsT = (unsigned short*)(w + 131072);     // 131072 B
  unsigned short* wxT = (unsigned short*)(w + 262144);     // 393216 B
  unsigned short* whT = (unsigned short*)(w + 655360);     // 393216 B
  unsigned short* pT  = (unsigned short*)(w + 1048576);    // 131072 B
  unsigned short* xg  = (unsigned short*)(w + 1179648);    // 2048*49*256*2 = 51380224 B
  float* state = (float*)(w + 52559872);                   // 2097152 B
  float* gruin = (float*)(w + 54657024);                   // 2097152 B

  k_transpose_bf16<<<(256 * 256 + 255) / 256, 256, 0, stream>>>(gat_kernel, wgT, 256, 256);
  k_transpose_bf16<<<(256 * 256 + 255) / 256, 256, 0, stream>>>(gat_self,   wsT, 256, 256);
  k_transpose_bf16<<<(256 * 768 + 255) / 256, 256, 0, stream>>>(gru_wx, wxT, 256, 768);
  k_transpose_bf16<<<(256 * 768 + 255) / 256, 256, 0, stream>>>(gru_wh, whT, 256, 768);
  k_transpose_bf16<<<(256 * 256 + 255) / 256, 256, 0, stream>>>(proj_w, pT, 256, 256);

  k_prep_x<<<GB, 256, 0, stream>>>(node_feature, xg, state);

  for (int s = 0; s < 4; ++s) {
    k_gat_step<<<GB, 256, 0, stream>>>(xg, wgT, wsT, att_src, att_dst, gruin);
    k_gru<<<GB / 16, 256, 0, stream>>>(gruin, state, wxT, whT, gru_bx, gru_bh, xg);
  }
  k_proj<<<GB / 16, 256, 0, stream>>>(state, pT, proj_b, (float*)d_out);

  (void)in_sizes; (void)n_in; (void)out_size; (void)ws_size;
}
